// GPTLanguageModel_41334765257078
// MI455X (gfx1250) — matrix operand order repaired and checked by reading, not compiled, not run on hardware
//
#include <hip/hip_runtime.h>
#include <hip/hip_bf16.h>
#include <math.h>

// ---------------------------------------------------------------------------
// GPT forward on gfx1250:
//   - f16 WMMA GEMMs (v_wmma_f32_16x16x32_f16), 64x128 tiles, 8 WMMA/wave/step
//   - all weights pre-transposed to [N][K] so A and B stage as contiguous b128
//   - async global->LDS staging via inline asm (ASYNCcnt), double-buffered LDS
//   - f32 residual stream, VALU online-softmax attention, fused epilogues
// ---------------------------------------------------------------------------

typedef __attribute__((ext_vector_type(16))) _Float16 v16h;
typedef __attribute__((ext_vector_type(8)))  _Float16 v8h;
typedef __attribute__((ext_vector_type(8)))  float    v8f;

// Flip to 0 if the async inline-asm path ever fails to assemble.
#define USE_ASYNC_LDS_ASM 1

#define B_  4096
#define T_  32
#define C_  384
#define H_  6
#define DH_ 64
#define L_  6
#define V_  80
#define C4_ (4 * C_)
#define M_  (B_ * T_)          // 131072 token rows

// 16-byte global -> LDS copy. Async path: GLOBAL_LOAD_ASYNC_TO_LDS_B128
// (tracked by ASYNCcnt). Flat pointers into the LDS aperture carry the
// wave-relative LDS byte address in their low 32 bits (ISA 10.2), so the
// truncated pointer is the VDST LDS-address operand.
__device__ __forceinline__ void cp16(const _Float16* g, _Float16* l) {
#if USE_ASYNC_LDS_ASM
  unsigned lofs = (unsigned)(size_t)l;
  asm volatile("global_load_async_to_lds_b128 %0, %1, off"
               :
               : "v"(lofs), "v"(g)
               : "memory");
#else
  *(v8h*)l = *(const v8h*)g;
#endif
}

__device__ __forceinline__ void wait_async() {
#if USE_ASYNC_LDS_ASM
  asm volatile("s_wait_asynccnt 0x0" ::: "memory");
#endif
}

// ------------------------------ embedding ----------------------------------
__global__ void k_embed(const int* __restrict__ idx, const float* __restrict__ tok,
                        const float* __restrict__ pos, float* __restrict__ x) {
  size_t i = (size_t)blockIdx.x * blockDim.x + threadIdx.x;
  size_t total = (size_t)M_ * C_;
  if (i >= total) return;
  int c   = (int)(i % C_);
  int row = (int)(i / C_);
  int t   = row % T_;
  int token = idx[row];
  x[i] = tok[(size_t)token * C_ + c] + pos[(size_t)t * C_ + c];
}

// --------------------------- weight converts -------------------------------
// Generic f32 [L][R][Cc] -> f16 transposed [L][Cc][R]
__global__ void k_transpose16(const float* __restrict__ in, _Float16* __restrict__ out,
                              int R, int Cc, size_t total) {
  size_t i = (size_t)blockIdx.x * blockDim.x + threadIdx.x;
  if (i >= total) return;
  size_t per = (size_t)R * Cc;
  int l  = (int)(i / per);
  int r2 = (int)(i % per);
  int c  = r2 / R;
  int r  = r2 % R;
  out[i] = (_Float16)in[(size_t)l * per + (size_t)r * Cc + c];
}

// Wq/Wk/Wv: [L,H,C,DH] -> f16 [L][N=h*DH+d][K=c]  (transposed for GEMM-B)
__global__ void k_qkv_perm(const float* __restrict__ in, _Float16* __restrict__ out, size_t n) {
  size_t i = (size_t)blockIdx.x * blockDim.x + threadIdx.x;
  if (i >= n) return;
  const int per = H_ * DH_ * C_;
  int l  = (int)(i / per);
  int r  = (int)(i % per);
  int nn = r / C_;
  int kk = r % C_;
  int h  = nn / DH_;
  int d  = nn % DH_;
  out[i] = (_Float16)in[(((size_t)l * H_ + h) * C_ + kk) * DH_ + d];
}

// ------------------------------ layernorm ----------------------------------
// one wave32 per row of 384; 4 rows per 128-thread block; f16 output
__global__ __launch_bounds__(128) void k_ln(const float* __restrict__ x,
                                            const float* __restrict__ g,
                                            const float* __restrict__ b,
                                            _Float16* __restrict__ out) {
  int wave = threadIdx.x >> 5;
  int lane = threadIdx.x & 31;
  int row  = blockIdx.x * 4 + wave;
  const float* xr = x + (size_t)row * C_;
  float vals[12];
  float s = 0.f;
#pragma unroll
  for (int i = 0; i < 12; ++i) { vals[i] = xr[lane + 32 * i]; s += vals[i]; }
#pragma unroll
  for (int off = 16; off; off >>= 1) s += __shfl_xor(s, off, 32);
  float mean = s * (1.0f / C_);
  float vv = 0.f;
#pragma unroll
  for (int i = 0; i < 12; ++i) { float d = vals[i] - mean; vv += d * d; }
#pragma unroll
  for (int off = 16; off; off >>= 1) vv += __shfl_xor(vv, off, 32);
  float inv = rsqrtf(vv * (1.0f / C_) + 1e-5f);
  _Float16* orow = out + (size_t)row * C_;
#pragma unroll
  for (int i = 0; i < 12; ++i) {
    int c = lane + 32 * i;
    orow[c] = (_Float16)((vals[i] - mean) * inv * g[c] + b[c]);
  }
}

// ------------------------------ WMMA GEMM ----------------------------------
// out[M,N] = A16[M,K] @ Bt16[N,K]^T (+bias) (+residual) (relu?) -> f32/f16
// Block: 128 threads (4 waves). Tile BM x TBN, K-step 32, double-buffered LDS
// staged with async global->LDS b128. Wave w owns rows [w*16, w*16+16).
#define BM 64
#define BK 32
template <int TBN, bool EDGE>
__global__ __launch_bounds__(128)
void k_gemm(const _Float16* __restrict__ A, const _Float16* __restrict__ Bt,
            const float* __restrict__ bias, const float* __restrict__ res,
            float* __restrict__ out32, _Float16* __restrict__ out16,
            int N, int K, int relu) {
  constexpr int NT = TBN / 16;                 // WMMAs per wave per K-step
  __shared__ _Float16 Asl[2][BM][BK];          // (m, k)
  __shared__ _Float16 Bsl[2][TBN][BK];         // (n, k): Bt is already [N][K]

  const int tid  = threadIdx.x;
  const int wave = tid >> 5;
  const int lane = tid & 31;
  const int hi   = lane >> 4;
  const int lo   = lane & 15;
  const int m0   = blockIdx.y * BM;
  const int n0   = blockIdx.x * TBN;

  auto stage = [&](int buf, int k0) {
    // A tile: BM*BK/8 = 256 chunks of 8 halves; 2 per thread
#pragma unroll
    for (int r = 0; r < (BM * BK / 8) / 128; ++r) {
      int ch  = tid + r * 128;
      int row = ch >> 2;
      int col = (ch & 3) << 3;
      cp16(&A[(size_t)(m0 + row) * K + k0 + col], &Asl[buf][row][col]);
    }
    // B tile: TBN*BK/8 chunks of 8 halves
#pragma unroll
    for (int r = 0; r < (TBN * BK / 8) / 128; ++r) {
      int ch  = tid + r * 128;
      int row = ch >> 2;
      int col = (ch & 3) << 3;
      if (EDGE) {
        int gn = n0 + row;
        v8h vb = {};
        if (gn < N) vb = *(const v8h*)&Bt[(size_t)gn * K + k0 + col];
        *(v8h*)&Bsl[buf][row][col] = vb;
      } else {
        cp16(&Bt[(size_t)(n0 + row) * K + k0 + col], &Bsl[buf][row][col]);
      }
    }
  };

  v8f acc[NT];
#pragma unroll
  for (int nt = 0; nt < NT; ++nt) acc[nt] = (v8f){};

  stage(0, 0);
  wait_async();
  __syncthreads();

  int buf = 0;
  for (int k0 = 0; k0 < K; k0 += BK) {
    if (k0 + BK < K) stage(buf ^ 1, k0 + BK);   // prefetch next tile (async)

    // A fragment (16x32 f16): lanes 0-15 K{0..7,16..23}, lanes 16-31 K{8..15,24..31}
    const _Float16* ar = &Asl[buf][wave * 16 + lo][0];
    v8h alo = *(const v8h*)(ar + hi * 8);
    v8h ahi = *(const v8h*)(ar + 16 + hi * 8);
    v16h af;
#pragma unroll
    for (int i = 0; i < 8; ++i) { af[i] = alo[i]; af[i + 8] = ahi[i]; }

    // B fragments (32x16 each): lane n = lo, lanes 0-15 K{0..7,16..23}, lanes 16-31 K{8..15,24..31}
#pragma unroll
    for (int nt = 0; nt < NT; ++nt) {
      const _Float16* br = &Bsl[buf][nt * 16 + lo][0];
      v8h blo = *(const v8h*)(br + hi * 8);
      v8h bhi = *(const v8h*)(br + 16 + hi * 8);
      v16h bf;
#pragma unroll
      for (int i = 0; i < 8; ++i) { bf[i] = blo[i]; bf[i + 8] = bhi[i]; }
      acc[nt] = __builtin_amdgcn_wmma_f32_16x16x32_f16(
          false, af, false, bf, (short)0, acc[nt], false, false);
    }

    wait_async();
    __syncthreads();
    buf ^= 1;
  }

  // epilogue: C/D layout: lane lo = col, VGPR r -> row r + 8*hi
#pragma unroll
  for (int nt = 0; nt < NT; ++nt) {
    int n = n0 + nt * 16 + lo;
    if (EDGE && n >= N) continue;
    float bv = bias ? bias[n] : 0.f;
#pragma unroll
    for (int r = 0; r < 8; ++r) {
      int m = m0 + wave * 16 + hi * 8 + r;
      float v = acc[nt][r] + bv;
      if (res)  v += res[(size_t)m * N + n];
      if (relu) v = fmaxf(v, 0.f);
      if (out32) out32[(size_t)m * N + n] = v;
      if (out16) out16[(size_t)m * N + n] = (_Float16)v;
    }
  }
}

// ------------------------------ attention ----------------------------------
// One wave32 per (b,h); lane = query row t; K,V resident in LDS; online softmax.
__global__ __launch_bounds__(32)
void k_attn(const _Float16* __restrict__ q, const _Float16* __restrict__ k,
            const _Float16* __restrict__ v, _Float16* __restrict__ o) {
  __shared__ float ks[T_][DH_];
  __shared__ float vs[T_][DH_];
  int bh = blockIdx.x;
  int b  = bh / H_;
  int h  = bh % H_;
  int lane = threadIdx.x;                       // t = lane
  size_t base = ((size_t)(b * T_ + lane)) * C_ + (size_t)h * DH_;
#pragma unroll
  for (int d = 0; d < DH_; ++d) {
    ks[lane][d] = (float)k[base + d];
    vs[lane][d] = (float)v[base + d];
  }
  __syncthreads();

  float qr[DH_];
#pragma unroll
  for (int d = 0; d < DH_; ++d) qr[d] = (float)q[base + d] * 0.125f;  // DH^-0.5

  float mrun = -3.0e38f, lsum = 0.f;
  float oacc[DH_];
#pragma unroll
  for (int d = 0; d < DH_; ++d) oacc[d] = 0.f;

  for (int s = 0; s <= lane; ++s) {             // causal
    float sc = 0.f;
#pragma unroll
    for (int d = 0; d < DH_; ++d) sc += qr[d] * ks[s][d];
    float nm   = fmaxf(mrun, sc);
    float corr = __expf(mrun - nm);
    float p    = __expf(sc - nm);
    lsum = lsum * corr + p;
#pragma unroll
    for (int d = 0; d < DH_; ++d) oacc[d] = oacc[d] * corr + p * vs[s][d];
    mrun = nm;
  }
  float inv = 1.f / lsum;
#pragma unroll
  for (int d = 0; d < DH_; ++d) o[base + d] = (_Float16)(oacc[d] * inv);
}

// ------------------------------- launch ------------------------------------
extern "C" void kernel_launch(void* const* d_in, const int* in_sizes, int n_in,
                              void* d_out, int out_size, void* d_ws, size_t ws_size,
                              hipStream_t stream) {
  (void)in_sizes; (void)n_in; (void)out_size; (void)ws_size;

  const int*   idx     = (const int*)  d_in[0];
  const float* tok_emb = (const float*)d_in[1];
  const float* pos_emb = (const float*)d_in[2];
  const float* ln1_g   = (const float*)d_in[3];
  const float* ln1_b   = (const float*)d_in[4];
  const float* Wq      = (const float*)d_in[5];
  const float* Wk      = (const float*)d_in[6];
  const float* Wv      = (const float*)d_in[7];
  const float* Wo      = (const float*)d_in[8];
  const float* bo      = (const float*)d_in[9];
  const float* ln2_g   = (const float*)d_in[10];
  const float* ln2_b   = (const float*)d_in[11];
  const float* W1      = (const float*)d_in[12];
  const float* b1      = (const float*)d_in[13];
  const float* W2      = (const float*)d_in[14];
  const float* b2      = (const float*)d_in[15];
  const float* lnf_g   = (const float*)d_in[16];
  const float* lnf_b   = (const float*)d_in[17];
  const float* Wlm     = (const float*)d_in[18];
  const float* blm     = (const float*)d_in[19];

  // ---- workspace layout ----
  char* p = (char*)d_ws;
  auto alloc = [&](size_t bytes) -> char* {
    char* r = p; p += (bytes + 255) & ~(size_t)255; return r;
  };
  float*    x     = (float*)   alloc((size_t)M_ * C_ * 4);
  _Float16* h16   = (_Float16*)alloc((size_t)M_ * C_ * 2);
  _Float16* q16   = (_Float16*)alloc((size_t)M_ * C_ * 2);
  _Float16* k16   = (_Float16*)alloc((size_t)M_ * C_ * 2);
  _Float16* v16_  = (_Float16*)alloc((size_t)M_ * C_ * 2);
  _Float16* o16   = (_Float16*)alloc((size_t)M_ * C_ * 2);
  _Float16* m16   = (_Float16*)alloc((size_t)M_ * C4_ * 2);
  _Float16* wq16  = (_Float16*)alloc((size_t)L_ * C_ * C_ * 2);   // [N][K]
  _Float16* wk16  = (_Float16*)alloc((size_t)L_ * C_ * C_ * 2);
  _Float16* wv16  = (_Float16*)alloc((size_t)L_ * C_ * C_ * 2);
  _Float16* wo16  = (_Float16*)alloc((size_t)L_ * C_ * C_ * 2);
  _Float16* w116  = (_Float16*)alloc((size_t)L_ * C4_ * C_ * 2);  // [4C][C]
  _Float16* w216  = (_Float16*)alloc((size_t)L_ * C_ * C4_ * 2);  // [C][4C]
  _Float16* wlm16 = (_Float16*)alloc((size_t)V_ * C_ * 2);        // [V][C]

  const int THR = 256;
  auto grid1 = [&](size_t n) { return dim3((unsigned)((n + THR - 1) / THR)); };

  // one-time weight conversion to f16, transposed to [N][K]
  size_t nqkv = (size_t)L_ * C_ * H_ * DH_;
  k_qkv_perm<<<grid1(nqkv), THR, 0, stream>>>(Wq, wq16, nqkv);
  k_qkv_perm<<<grid1(nqkv), THR, 0, stream>>>(Wk, wk16, nqkv);
  k_qkv_perm<<<grid1(nqkv), THR, 0, stream>>>(Wv, wv16, nqkv);
  size_t nwo = (size_t)L_ * C_ * C_;
  size_t nw1 = (size_t)L_ * C_ * C4_;
  k_transpose16<<<grid1(nwo), THR, 0, stream>>>(Wo, wo16, C_,  C_,  nwo);
  k_transpose16<<<grid1(nw1), THR, 0, stream>>>(W1, w116, C_,  C4_, nw1);
  k_transpose16<<<grid1(nw1), THR, 0, stream>>>(W2, w216, C4_, C_,  nw1);
  { size_t n = (size_t)C_ * V_;
    k_transpose16<<<grid1(n), THR, 0, stream>>>(Wlm, wlm16, C_, V_, n); }

  // embedding
  k_embed<<<grid1((size_t)M_ * C_), THR, 0, stream>>>(idx, tok_emb, pos_emb, x);

  const dim3 blk(128);
  const dim3 lnG(M_ / 4);
  const dim3 gC (C_  / 128, M_ / BM);             // (3,  2048)
  const dim3 gC4(C4_ / 128, M_ / BM);             // (12, 2048)
  const dim3 gV ((V_ + 63) / 64, M_ / BM);        // (2,  2048), edge path

  for (int l = 0; l < L_; ++l) {
    k_ln<<<lnG, blk, 0, stream>>>(x, ln1_g + (size_t)l * C_, ln1_b + (size_t)l * C_, h16);
    k_gemm<128, false><<<gC, blk, 0, stream>>>(h16, wq16 + (size_t)l * C_ * C_,
                                               nullptr, nullptr, nullptr, q16, C_, C_, 0);
    k_gemm<128, false><<<gC, blk, 0, stream>>>(h16, wk16 + (size_t)l * C_ * C_,
                                               nullptr, nullptr, nullptr, k16, C_, C_, 0);
    k_gemm<128, false><<<gC, blk, 0, stream>>>(h16, wv16 + (size_t)l * C_ * C_,
                                               nullptr, nullptr, nullptr, v16_, C_, C_, 0);
    k_attn<<<dim3(B_ * H_), dim3(32), 0, stream>>>(q16, k16, v16_, o16);
    k_gemm<128, false><<<gC, blk, 0, stream>>>(o16, wo16 + (size_t)l * C_ * C_,
                                               bo + (size_t)l * C_, x, x, nullptr, C_, C_, 0);
    k_ln<<<lnG, blk, 0, stream>>>(x, ln2_g + (size_t)l * C_, ln2_b + (size_t)l * C_, h16);
    k_gemm<128, false><<<gC4, blk, 0, stream>>>(h16, w116 + (size_t)l * C_ * C4_,
                                                b1 + (size_t)l * C4_, nullptr, nullptr, m16,
                                                C4_, C_, 1 /*relu*/);
    k_gemm<128, false><<<gC, blk, 0, stream>>>(m16, w216 + (size_t)l * C4_ * C_,
                                               b2 + (size_t)l * C_, x, x, nullptr, C_, C4_, 0);
  }

  k_ln<<<lnG, blk, 0, stream>>>(x, lnf_g, lnf_b, h16);
  k_gemm<64, true><<<gV, blk, 0, stream>>>(h16, wlm16, blm, nullptr,
                                           (float*)d_out, nullptr, V_, C_, 0);
}
